// ResMambaBlock_45621142618613
// MI455X (gfx1250) — matrix operand order repaired and checked by reading, not compiled, not run on hardware
//
#include <hip/hip_runtime.h>
#include <hip/hip_bf16.h>
#include <math.h>

typedef __attribute__((ext_vector_type(16))) _Float16 v16h;
typedef __attribute__((ext_vector_type(8)))  float    v8f;

#define N_FEATS 64
#define D_INNER 128
#define D_STATE 64
#define DT_RANK 4
#define HIDDEN  256
#define BATCH   2
#define LTOK    4096            // H*W
#define MROWS   8192            // B*L
#define LDDBL   132             // DT_RANK + 2*D_STATE
#define NCHUNK  64              // scan chunks
#define CLEN    64              // tokens per chunk
#define KMAX    256             // largest GEMM K

// ---------------------------------------------------------------------------
// WMMA fragment helpers (f32 -> f16 on load, f32 accumulate)
// A: 16x32 tile of row-major [M,K] fp32.  Lane layout per ISA 7.12.2:
//   m = lane&15 ; koff = (lane>>4)*8 ; elems 0..7 -> k0+koff+e ; 8..15 -> k0+16+koff+(e-8)
// ---------------------------------------------------------------------------
__device__ __forceinline__ void pack8(v16h& v, int base, const float4 f0, const float4 f1) {
    v[base + 0] = (_Float16)f0.x; v[base + 1] = (_Float16)f0.y;
    v[base + 2] = (_Float16)f0.z; v[base + 3] = (_Float16)f0.w;
    v[base + 4] = (_Float16)f1.x; v[base + 5] = (_Float16)f1.y;
    v[base + 6] = (_Float16)f1.z; v[base + 7] = (_Float16)f1.w;
}

__device__ __forceinline__ v16h load_a_frag(const float* __restrict__ A,
                                            int m0, int k0, int K, int lane) {
    const float* p = A + (size_t)(m0 + (lane & 15)) * K + k0 + ((lane >> 4) << 3);
    float4 f0 = *(const float4*)(p);
    float4 f1 = *(const float4*)(p + 4);
    float4 f2 = *(const float4*)(p + 16);
    float4 f3 = *(const float4*)(p + 20);
    v16h a;
    pack8(a, 0, f0, f1);
    pack8(a, 8, f2, f3);
    return a;
}

// B fragment from the LDS-staged weight slab (row-major [rows][K] fp32).
//   nloc = column tile within slab ; koff = (lane>>4)*8 ; elems 0..7 -> k0+koff+e ; 8..15 -> k0+16+koff+(e-8)
__device__ __forceinline__ v16h load_b_frag_lds(const float* bsh,
                                                int nloc, int k0, int K,
                                                int validRows, int lane) {
    v16h b = {};
    int n = nloc + (lane & 15);
    if (n < validRows) {
        const float* p = bsh + (size_t)n * K + k0 + ((lane >> 4) << 3);
        float4 f0 = *(const float4*)(p);
        float4 f1 = *(const float4*)(p + 4);
        float4 f2 = *(const float4*)(p + 16);
        float4 f3 = *(const float4*)(p + 20);
        pack8(b, 0, f0, f1);
        pack8(b, 8, f2, f3);
    }
    return b;
}

// C/D 16x16 f32: n = lane&15 ; row r of v8f -> m = m0 + (lane>>4)*8 + r
__device__ __forceinline__ void store_c_frag(float* __restrict__ out, const v8f c,
                                             int m0, int n0, int N, int ldout,
                                             const float* __restrict__ bias,
                                             int accumulate, int lane) {
    int n = n0 + (lane & 15);
    if (n >= N) return;
    float bv = bias ? bias[n] : 0.0f;
    int mbase = m0 + ((lane >> 4) << 3);
    #pragma unroll
    for (int r = 0; r < 8; ++r) {
        float v = c[r] + bv;
        size_t idx = (size_t)(mbase + r) * ldout + n;
        if (accumulate) out[idx] += v; else out[idx] = v;
    }
}

// ---------------------------------------------------------------------------
// Generic GEMM: out[M,ldout] (+)= A[M,K] * Wt[N,K]^T (+ bias)
// Block = 256 threads = 8 waves; wave tile 32x32 (2x2 WMMA frags);
// block tile 128(M) x 64(N).  M, K multiples of 32; N may be ragged.
// The block's 64-row weight slab is staged into LDS once via the gfx1250
// async global->LDS path (ASYNCcnt), then B fragments come from LDS.
// ---------------------------------------------------------------------------
__global__ void gemm_wmma_f16(const float* __restrict__ A,
                              const float* __restrict__ Wt,
                              const float* __restrict__ bias,
                              float* __restrict__ out,
                              int N, int K, int ldout, int accumulate) {
    __shared__ float bsh[64 * KMAX];     // 64KB max (K <= 256)

    const int lane = threadIdx.x & 31;
    const int wave = threadIdx.x >> 5;
    const int m0    = blockIdx.x * 128 + (wave >> 1) * 32;
    const int nBase = blockIdx.y * 64;
    const int nloc0 = (wave & 1) * 32;

    // ---- async-stage weight slab: Wt rows [nBase, nBase+validRows) are a
    // contiguous validRows*K-float region (row-major [N,K]) ----
    const int validRows = (N - nBase < 64) ? (N - nBase) : 64;
    {
        const float* src = Wt + (size_t)nBase * K;
        const unsigned ldsBase = (unsigned)(size_t)(&bsh[0]);   // LDS aperture: addr[31:0] = LDS offset
        const int nVec = (validRows * K) >> 2;                  // float4 transfers
        for (int i = threadIdx.x; i < nVec; i += 256) {
            unsigned ldsAddr = ldsBase + (unsigned)(i << 4);
            const float* g = src + ((size_t)i << 2);
            asm volatile("global_load_async_to_lds_b128 %0, %1, off"
                         :: "v"(ldsAddr), "v"(g) : "memory");
        }
        asm volatile("s_wait_asynccnt 0x0" ::: "memory");
    }
    __syncthreads();

    v8f c00 = {}, c01 = {}, c10 = {}, c11 = {};

    for (int k0 = 0; k0 < K; k0 += 32) {
        if (k0 + 32 < K) {   // gfx1250 global_prefetch_b8 for the next A slab
            __builtin_prefetch(A + (size_t)(m0 + (lane & 15)) * K + k0 + 32, 0, 3);
        }
        v16h a0 = load_a_frag(A, m0,      k0, K, lane);
        v16h a1 = load_a_frag(A, m0 + 16, k0, K, lane);
        v16h b0 = load_b_frag_lds(bsh, nloc0,      k0, K, validRows, lane);
        v16h b1 = load_b_frag_lds(bsh, nloc0 + 16, k0, K, validRows, lane);
        c00 = __builtin_amdgcn_wmma_f32_16x16x32_f16(false, a0, false, b0, (short)0, c00, false, false);
        c01 = __builtin_amdgcn_wmma_f32_16x16x32_f16(false, a0, false, b1, (short)0, c01, false, false);
        c10 = __builtin_amdgcn_wmma_f32_16x16x32_f16(false, a1, false, b0, (short)0, c10, false, false);
        c11 = __builtin_amdgcn_wmma_f32_16x16x32_f16(false, a1, false, b1, (short)0, c11, false, false);
    }

    const int n0 = nBase + nloc0;
    store_c_frag(out, c00, m0,      n0,      N, ldout, bias, accumulate, lane);
    store_c_frag(out, c01, m0,      n0 + 16, N, ldout, bias, accumulate, lane);
    store_c_frag(out, c10, m0 + 16, n0,      N, ldout, bias, accumulate, lane);
    store_c_frag(out, c11, m0 + 16, n0 + 16, N, ldout, bias, accumulate, lane);
}

// ---------------------------------------------------------------------------
// Direct 3x3 conv, C=64 -> 64, SAME padding. One block per (c_out, batch).
// token_layout=1 stores out[b][h*W+w][64] (channel-last tokens).
// ---------------------------------------------------------------------------
__global__ void conv3x3_kernel(const float* __restrict__ x,
                               const float* __restrict__ w,
                               const float* __restrict__ bias,
                               float* __restrict__ out,
                               int do_relu, int token_layout) {
    const int co = blockIdx.x;
    const int b  = blockIdx.y;
    __shared__ float wsh[64 * 9];
    for (int i = threadIdx.x; i < 576; i += 256) wsh[i] = w[(size_t)co * 576 + i];
    __syncthreads();

    const float bv = bias[co];
    const float* xb = x + (size_t)b * 64 * LTOK;

    for (int p = threadIdx.x; p < LTOK; p += 256) {
        const int h = p >> 6, wc = p & 63;
        float acc = bv;
        for (int ci = 0; ci < 64; ++ci) {
            const float* xc = xb + (size_t)ci * LTOK;
            const float* wk = wsh + ci * 9;
            #pragma unroll
            for (int ky = 0; ky < 3; ++ky) {
                const int hy = h + ky - 1;
                if (hy < 0 || hy > 63) continue;
                #pragma unroll
                for (int kx = 0; kx < 3; ++kx) {
                    const int wx = wc + kx - 1;
                    if (wx < 0 || wx > 63) continue;
                    acc = fmaf(xc[hy * 64 + wx], wk[ky * 3 + kx], acc);
                }
            }
        }
        if (do_relu) acc = fmaxf(acc, 0.0f);
        if (token_layout)
            out[(size_t)b * LTOK * 64 + (size_t)p * 64 + co] = acc;
        else
            out[(size_t)b * 64 * LTOK + (size_t)co * LTOK + p] = acc;
    }
}

// ---------------------------------------------------------------------------
// LayerNorm over rows of 64; one wave per row (2 elems/lane under wave32).
// ---------------------------------------------------------------------------
__global__ void layernorm64(const float* __restrict__ in,
                            const float* __restrict__ w,
                            const float* __restrict__ b,
                            float* __restrict__ out, int rows) {
    const int row  = blockIdx.x * (blockDim.x >> 5) + (threadIdx.x >> 5);
    const int lane = threadIdx.x & 31;
    if (row >= rows) return;
    const float* p = in + (size_t)row * 64;
    float x0 = p[lane], x1 = p[lane + 32];
    float s = x0 + x1;
    #pragma unroll
    for (int o = 16; o > 0; o >>= 1) s += __shfl_xor(s, o, 32);
    const float mean = s * (1.0f / 64.0f);
    const float d0 = x0 - mean, d1 = x1 - mean;
    float v = d0 * d0 + d1 * d1;
    #pragma unroll
    for (int o = 16; o > 0; o >>= 1) v += __shfl_xor(v, o, 32);
    const float rstd = rsqrtf(v * (1.0f / 64.0f) + 1e-5f);
    out[(size_t)row * 64 + lane]      = d0 * rstd * w[lane]      + b[lane];
    out[(size_t)row * 64 + lane + 32] = d1 * rstd * w[lane + 32] + b[lane + 32];
}

// ---------------------------------------------------------------------------
// Causal depthwise conv1d (window 4) along L on xin = xz[...,0:128], + SiLU.
// ---------------------------------------------------------------------------
__global__ void conv1d_silu(const float* __restrict__ xz,
                            const float* __restrict__ w,
                            const float* __restrict__ bias,
                            float* __restrict__ xc) {
    const int idx = blockIdx.x * blockDim.x + threadIdx.x;   // B*L*128 = 2^20
    if (idx >= BATCH * LTOK * D_INNER) return;
    const int d = idx & 127;
    const int l = (idx >> 7) & (LTOK - 1);
    const int b = idx >> 19;
    const float* base = xz + (size_t)b * LTOK * 256;
    float acc = bias[d];
    #pragma unroll
    for (int j = 0; j < 4; ++j) {
        const int ll = l - 3 + j;
        if (ll >= 0) acc = fmaf(base[(size_t)ll * 256 + d], w[d * 4 + j], acc);
    }
    acc = acc / (1.0f + __expf(-acc));     // silu
    xc[((size_t)b * LTOK + l) * D_INNER + d] = acc;
}

// ---------------------------------------------------------------------------
// delta = softplus(dt @ dt_proj_w.T + dt_proj_b); dt = dbl[...,0:4]
// ---------------------------------------------------------------------------
__global__ void dt_softplus(const float* __restrict__ dbl,
                            const float* __restrict__ dtw,
                            const float* __restrict__ dtb,
                            float* __restrict__ delta) {
    const int idx = blockIdx.x * blockDim.x + threadIdx.x;
    if (idx >= MROWS * D_INNER) return;
    const int d = idx & 127;
    const int row = idx >> 7;
    float acc = dtb[d];
    #pragma unroll
    for (int r = 0; r < DT_RANK; ++r)
        acc = fmaf(dbl[(size_t)row * LDDBL + r], dtw[d * DT_RANK + r], acc);
    delta[idx] = (acc > 20.0f) ? acc : log1pf(__expf(acc));
}

// ---------------------------------------------------------------------------
// Chunked parallel selective scan.
//   h_l = dA_l * h_{l-1} + (dt_l*u_l)*B_l   (elementwise per (b,d,n))
// Phase 1: per (b,d,chunk) compute transfer fn  h_out = P*h_in + Q   (h_in=0)
// Phase 2: per (b,d,n) sequential combine over 64 chunks -> chunk-entry Hs
// Phase 3: per (b,d,chunk) replay from Hs, emit y_l = sum_n h*C + u*D
// One wave per (d,chunk): 64 states as 2/lane under wave32.
// ---------------------------------------------------------------------------
__global__ void scan_phase1(const float* __restrict__ delta,
                            const float* __restrict__ xc,
                            const float* __restrict__ dbl,
                            const float* __restrict__ A_log,
                            float* __restrict__ P, float* __restrict__ Q) {
    const int b     = blockIdx.z;
    const int chunk = blockIdx.y;
    const int d     = blockIdx.x * 8 + (threadIdx.x >> 5);
    const int lane  = threadIdx.x & 31;
    const int n0 = lane, n1 = lane + 32;

    const float a0 = -__expf(A_log[(size_t)d * D_STATE + n0]);
    const float a1 = -__expf(A_log[(size_t)d * D_STATE + n1]);

    const int l0 = chunk * CLEN;
    const float* del = delta + (size_t)b * LTOK * D_INNER + d;
    const float* u   = xc    + (size_t)b * LTOK * D_INNER + d;
    const float* Bp  = dbl   + (size_t)b * LTOK * LDDBL + DT_RANK;

    float p0 = 1.0f, p1 = 1.0f, q0 = 0.0f, q1 = 0.0f;
    for (int l = l0; l < l0 + CLEN; ++l) {
        const float dt = del[(size_t)l * D_INNER];
        const float ut = u[(size_t)l * D_INNER];
        const float Bv0 = Bp[(size_t)l * LDDBL + n0];
        const float Bv1 = Bp[(size_t)l * LDDBL + n1];
        const float dA0 = __expf(dt * a0);
        const float dA1 = __expf(dt * a1);
        const float du = dt * ut;
        q0 = fmaf(q0, dA0, du * Bv0);
        q1 = fmaf(q1, dA1, du * Bv1);
        p0 *= dA0;
        p1 *= dA1;
    }
    const size_t base = (((size_t)b * NCHUNK + chunk) * D_INNER + d) * D_STATE;
    P[base + n0] = p0; P[base + n1] = p1;
    Q[base + n0] = q0; Q[base + n1] = q1;
}

__global__ void scan_phase2(const float* __restrict__ P,
                            const float* __restrict__ Q,
                            float* __restrict__ Hs) {
    const int idx = blockIdx.x * blockDim.x + threadIdx.x;   // B*128*64 = 16384
    if (idx >= BATCH * D_INNER * D_STATE) return;
    const int n = idx & 63;
    const int d = (idx >> 6) & 127;
    const int b = idx >> 13;
    float h = 0.0f;
    for (int c = 0; c < NCHUNK; ++c) {
        const size_t o = (((size_t)b * NCHUNK + c) * D_INNER + d) * D_STATE + n;
        Hs[o] = h;                       // state entering chunk c
        h = fmaf(P[o], h, Q[o]);
    }
}

__global__ void scan_phase3(const float* __restrict__ delta,
                            const float* __restrict__ xc,
                            const float* __restrict__ dbl,
                            const float* __restrict__ A_log,
                            const float* __restrict__ Dp,
                            const float* __restrict__ Hs,
                            float* __restrict__ ys) {
    const int b     = blockIdx.z;
    const int chunk = blockIdx.y;
    const int d     = blockIdx.x * 8 + (threadIdx.x >> 5);
    const int lane  = threadIdx.x & 31;
    const int n0 = lane, n1 = lane + 32;

    const float a0 = -__expf(A_log[(size_t)d * D_STATE + n0]);
    const float a1 = -__expf(A_log[(size_t)d * D_STATE + n1]);
    const float Dd = Dp[d];

    const size_t base = (((size_t)b * NCHUNK + chunk) * D_INNER + d) * D_STATE;
    float h0 = Hs[base + n0];
    float h1 = Hs[base + n1];

    const int l0 = chunk * CLEN;
    const float* del = delta + (size_t)b * LTOK * D_INNER + d;
    const float* u   = xc    + (size_t)b * LTOK * D_INNER + d;
    const float* Bp  = dbl   + (size_t)b * LTOK * LDDBL + DT_RANK;
    const float* Cp  = dbl   + (size_t)b * LTOK * LDDBL + DT_RANK + D_STATE;
    float* yo        = ys    + (size_t)b * LTOK * D_INNER + d;

    for (int l = l0; l < l0 + CLEN; ++l) {
        const float dt = del[(size_t)l * D_INNER];
        const float ut = u[(size_t)l * D_INNER];
        const float Bv0 = Bp[(size_t)l * LDDBL + n0];
        const float Bv1 = Bp[(size_t)l * LDDBL + n1];
        const float Cv0 = Cp[(size_t)l * LDDBL + n0];
        const float Cv1 = Cp[(size_t)l * LDDBL + n1];
        const float du = dt * ut;
        h0 = fmaf(h0, __expf(dt * a0), du * Bv0);
        h1 = fmaf(h1, __expf(dt * a1), du * Bv1);
        float y = h0 * Cv0 + h1 * Cv1;
        #pragma unroll
        for (int o = 16; o > 0; o >>= 1) y += __shfl_xor(y, o, 32);
        if (lane == 0) yo[(size_t)l * D_INNER] = y + ut * Dd;
    }
}

// ---------------------------------------------------------------------------
// g = ys * silu(z), z = xz[...,128:256]
// ---------------------------------------------------------------------------
__global__ void gate_silu(const float* __restrict__ ys,
                          const float* __restrict__ xz,
                          float* __restrict__ g) {
    const int idx = blockIdx.x * blockDim.x + threadIdx.x;
    if (idx >= MROWS * D_INNER) return;
    const int d = idx & 127;
    const int row = idx >> 7;
    const float z = xz[(size_t)row * 256 + 128 + d];
    g[idx] = ys[idx] * (z / (1.0f + __expf(-z)));
}

// ---------------------------------------------------------------------------
// Depthwise 3x3 over hmid tokens [b][l][256] viewed as [b][256][64][64]; + GELU.
// ---------------------------------------------------------------------------
__global__ void dwconv_gelu(const float* __restrict__ hm,
                            const float* __restrict__ w,
                            const float* __restrict__ bias,
                            float* __restrict__ out) {
    const int idx = blockIdx.x * blockDim.x + threadIdx.x;  // 2^21
    if (idx >= BATCH * LTOK * HIDDEN) return;
    const int ch = idx & 255;
    const int p  = (idx >> 8) & (LTOK - 1);
    const int b  = idx >> 20;
    const int h = p >> 6, wc = p & 63;
    const float* base = hm + (size_t)b * LTOK * HIDDEN + ch;
    float acc = bias[ch];
    #pragma unroll
    for (int ky = 0; ky < 3; ++ky) {
        const int hy = h + ky - 1;
        if (hy < 0 || hy > 63) continue;
        #pragma unroll
        for (int kx = 0; kx < 3; ++kx) {
            const int wx = wc + kx - 1;
            if (wx < 0 || wx > 63) continue;
            acc = fmaf(base[(size_t)(hy * 64 + wx) * HIDDEN], w[ch * 9 + ky * 3 + kx], acc);
        }
    }
    out[idx] = 0.5f * acc * (1.0f + erff(acc * 0.70710678118654752f));   // exact gelu
}

// ---------------------------------------------------------------------------
// out[b][c][h][w] = x[b][c][h][w] + t[b][h*W+w][c]
// ---------------------------------------------------------------------------
__global__ void final_add(const float* __restrict__ x,
                          const float* __restrict__ t,
                          float* __restrict__ out) {
    const int idx = blockIdx.x * blockDim.x + threadIdx.x;   // 2^19
    if (idx >= BATCH * 64 * LTOK) return;
    const int p = idx & (LTOK - 1);
    const int c = (idx >> 12) & 63;
    const int b = idx >> 18;
    out[idx] = x[idx] + t[((size_t)b * LTOK + p) * 64 + c];
}

// ---------------------------------------------------------------------------
extern "C" void kernel_launch(void* const* d_in, const int* in_sizes, int n_in,
                              void* d_out, int out_size, void* d_ws, size_t ws_size,
                              hipStream_t stream) {
    (void)in_sizes; (void)n_in; (void)out_size; (void)ws_size;
    const float* x         = (const float*)d_in[0];
    const float* conv1_w   = (const float*)d_in[1];
    const float* conv1_b   = (const float*)d_in[2];
    const float* conv2_w   = (const float*)d_in[3];
    const float* conv2_b   = (const float*)d_in[4];
    const float* ln1_w     = (const float*)d_in[5];
    const float* ln1_b     = (const float*)d_in[6];
    const float* ln2_w     = (const float*)d_in[7];
    const float* ln2_b     = (const float*)d_in[8];
    const float* in_proj_w = (const float*)d_in[9];
    const float* conv1d_w  = (const float*)d_in[10];
    const float* conv1d_b  = (const float*)d_in[11];
    const float* x_proj_w  = (const float*)d_in[12];
    const float* dt_proj_w = (const float*)d_in[13];
    const float* dt_proj_b = (const float*)d_in[14];
    const float* A_log     = (const float*)d_in[15];
    const float* D_param   = (const float*)d_in[16];
    const float* out_proj_w= (const float*)d_in[17];
    const float* fc1_w     = (const float*)d_in[18];
    const float* fc1_b     = (const float*)d_in[19];
    const float* dw_w      = (const float*)d_in[20];
    const float* dw_b      = (const float*)d_in[21];
    const float* fc2_w     = (const float*)d_in[22];
    const float* fc2_b     = (const float*)d_in[23];
    float* out = (float*)d_out;

    float* ws    = (float*)d_ws;
    float* t     = ws;                 //  524288  tokens [B][L][64]
    float* ln    = ws + 524288;        //  524288  LN output (ln1 then ln2)
    float* xz    = ws + 1048576;       // 2097152  in_proj out; later gelu(dwconv) out
    float* xc    = ws + 3145728;       // 1048576  conv1d+silu out
    float* dbl   = ws + 4194304;       // 1081344  x_proj out [M][132]
    float* delta = ws + 5275648;       // 1048576  softplus(dt); reused as gated g
    float* ys    = ws + 6324224;       // 1048576  scan output
    float* hmid  = ws + 7372800;       // 2097152  fc1 out (written late)
    float* Hs    = ws + 9469952;       // 1048576  chunk-entry scan states
    float* y1    = hmid;               // conv1 intermediate reuses hmid slot
    float* P     = hmid;               // scan transfer-fn products reuse hmid
    float* Q     = hmid + 1048576;     //   (both dead before fc1 writes hmid)

    // conv1 + relu (NCHW), conv2 writing token layout [B][L][64]
    conv3x3_kernel<<<dim3(64, 2), 256, 0, stream>>>(x,  conv1_w, conv1_b, y1, 1, 0);
    conv3x3_kernel<<<dim3(64, 2), 256, 0, stream>>>(y1, conv2_w, conv2_b, t,  0, 1);

    // ----- Mamba -----
    layernorm64<<<1024, 256, 0, stream>>>(t, ln1_w, ln1_b, ln, MROWS);
    gemm_wmma_f16<<<dim3(64, 4), 256, 0, stream>>>(ln, in_proj_w, nullptr, xz, 256, 64, 256, 0);
    conv1d_silu<<<4096, 256, 0, stream>>>(xz, conv1d_w, conv1d_b, xc);
    gemm_wmma_f16<<<dim3(64, 3), 256, 0, stream>>>(xc, x_proj_w, nullptr, dbl, LDDBL, 128, LDDBL, 0);
    dt_softplus<<<4096, 256, 0, stream>>>(dbl, dt_proj_w, dt_proj_b, delta);

    // chunked parallel selective scan (64 chunks of 64)
    scan_phase1<<<dim3(16, NCHUNK, BATCH), 256, 0, stream>>>(delta, xc, dbl, A_log, P, Q);
    scan_phase2<<<64, 256, 0, stream>>>(P, Q, Hs);
    scan_phase3<<<dim3(16, NCHUNK, BATCH), 256, 0, stream>>>(delta, xc, dbl, A_log, D_param, Hs, ys);

    gate_silu<<<4096, 256, 0, stream>>>(ys, xz, delta);   // delta reused as g
    gemm_wmma_f16<<<dim3(64, 1), 256, 0, stream>>>(delta, out_proj_w, nullptr, t, 64, 128, 64, 1); // t += out_proj(g)

    // ----- PVT2 FFN -----
    layernorm64<<<1024, 256, 0, stream>>>(t, ln2_w, ln2_b, ln, MROWS);
    gemm_wmma_f16<<<dim3(64, 4), 256, 0, stream>>>(ln, fc1_w, fc1_b, hmid, 256, 64, 256, 0);
    dwconv_gelu<<<8192, 256, 0, stream>>>(hmid, dw_w, dw_b, xz);  // xz reused as activated hidden
    gemm_wmma_f16<<<dim3(64, 1), 256, 0, stream>>>(xz, fc2_w, fc2_b, t, 64, 256, 64, 1); // t += fc2(...)

    // residual + layout back to NCHW
    final_add<<<2048, 256, 0, stream>>>(x, t, out);
}
